// NeuralODEModel_10943576670566
// MI455X (gfx1250) — matrix operand order repaired and checked by reading, not compiled, not run on hardware
//
#include <hip/hip_runtime.h>

typedef __attribute__((ext_vector_type(16))) _Float16 v16h;
typedef __attribute__((ext_vector_type(8)))  float    v8f;

// gfx1250 hardware tanh (TRANS32) if the toolchain exposes it; else exp2/rcp.
__device__ __forceinline__ float fast_tanh(float x) {
#if __has_builtin(__builtin_amdgcn_tanhf)
    return __builtin_amdgcn_tanhf(x);
#elif __has_builtin(__builtin_amdgcn_tanh_f32)
    return __builtin_amdgcn_tanh_f32(x);
#else
    // tanh(x) = 1 - 2/(exp(2x)+1), exp(2x) = exp2(x * 2*log2(e))
    float e = __builtin_amdgcn_exp2f(x * 2.885390081777927f);
    return 1.0f - 2.0f * __builtin_amdgcn_rcpf(e + 1.0f);
#endif
}

// One wave (32 lanes) integrates 16 independent scalar ODE chains.
// Sample n (= lane & 15) is owned redundantly by lanes n and n+16:
//   - B-matrix (32x16 f16): lane holds its sample's hidden column,
//     lanes 0-15 hold K=0..15, lanes 16-31 hold K=16..31 (per K-chunk of 32).
//   - A-matrix (16x32 f16): W2 replicated in every row -> every row of D equals
//     the per-sample output vector, so each lane's D VGPR0 element is exactly
//     its own sample's f(y). No cross-lane traffic in the serial RK chain.
__global__ __launch_bounds__(256) void node_rk4_wmma_kernel(
    const float* __restrict__ x,  const float* __restrict__ W1,
    const float* __restrict__ b1, const float* __restrict__ W2,
    const float* __restrict__ b2p, float* __restrict__ out,
    int seq_len)
{
    const int t    = blockIdx.x * blockDim.x + threadIdx.x;
    const int lane = t & 31;
    const int half = lane >> 4;       // which K-half of B this lane supplies
    const int n    = lane & 15;       // sample column within the wave tile
    const int s    = (t >> 5) * 16 + n; // global sample id

    // Per-lane layer-1 weights for the B rows this lane produces:
    //   chunk c in {0,1}: global k = 32*c + 16*half + e,  e = 0..15
    float w1v[32], b1v[32];
    #pragma unroll
    for (int c = 0; c < 2; ++c) {
        #pragma unroll
        for (int e = 0; e < 16; ++e) {
            int k = 32 * c + 16 * half + e;
            bool ok = (k < 50);
            w1v[c * 16 + e] = ok ? W1[k] : 0.0f;   // pad K 50..63 with zeros
            b1v[c * 16 + e] = ok ? b1[k] : 0.0f;
        }
    }

    // A matrices (one per K-chunk): W2 replicated across all 16 rows.
    // A uses the same contiguous k order as B: lanes 0-15 element e -> K = e,
    // lanes 16-31 -> K = 16 + e. Rows identical => lane-half only.
    v16h a0, a1;
    #pragma unroll
    for (int e = 0; e < 16; ++e) {
        int off = half * 16 + e;
        int k0 = off, k1 = 32 + off;
        a0[e] = (_Float16)((k0 < 50) ? W2[k0] : 0.0f);
        a1[e] = (_Float16)((k1 < 50) ? W2[k1] : 0.0f);
    }

    const float b2 = b2p[0];

    // MLP vector field: both K-chunks reduced through WMMA with f32 accum.
    auto feval = [&](float yy) -> float {
        v8f c = {};
        v16h bm;
        #pragma unroll
        for (int e = 0; e < 16; ++e)
            bm[e] = (_Float16)fast_tanh(yy * w1v[e] + b1v[e]);
        c = __builtin_amdgcn_wmma_f32_16x16x32_f16(
                false, a0, false, bm, (short)0, c, false, false);
        #pragma unroll
        for (int e = 0; e < 16; ++e)
            bm[e] = (_Float16)fast_tanh(yy * w1v[16 + e] + b1v[16 + e]);
        c = __builtin_amdgcn_wmma_f32_16x16x32_f16(
                false, a1, false, bm, (short)0, c, false, false);
        return c[0] + b2;   // every lane reads its own sample's result
    };

    // y0 = x[s, seq_len-1, 0]; strided one-time load (negligible traffic).
    float y = x[(size_t)s * (size_t)seq_len + (size_t)(seq_len - 1)];

    const float dt    = 1.0f / (float)(seq_len - 1);
    const float third = 1.0f / 3.0f;

    for (int step = 0; step < seq_len - 1; ++step) {
        float k1 = feval(y);
        float k2 = feval(y + dt * k1 * third);
        float k3 = feval(y + dt * (k2 - k1 * third));
        float k4 = feval(y + dt * (k1 - k2 + k3));
        y = y + (k1 + 3.0f * (k2 + k3) + k4) * (dt * 0.125f);
    }

    // Only divergence in the kernel: one of the two redundant lanes stores.
    if (half == 0)
        out[s] = y;
}

extern "C" void kernel_launch(void* const* d_in, const int* in_sizes, int n_in,
                              void* d_out, int out_size, void* d_ws, size_t ws_size,
                              hipStream_t stream) {
    const float* x  = (const float*)d_in[0];
    const float* W1 = (const float*)d_in[1];
    const float* b1 = (const float*)d_in[2];
    const float* W2 = (const float*)d_in[3];
    const float* b2 = (const float*)d_in[4];
    float* out = (float*)d_out;

    const int batch   = out_size;               // 65536
    const int seq_len = in_sizes[0] / batch;    // 1000

    const int threads = batch * 2;              // 2 lanes per sample
    dim3 block(256);
    dim3 grid((threads + 255) / 256);
    node_rk4_wmma_kernel<<<grid, block, 0, stream>>>(x, W1, b1, W2, b2, out, seq_len);
}
